// StructuredStateTable_369367187860
// MI455X (gfx1250) — compile-verified, matrix operand order checked by reading, not run on hardware
//
#include <hip/hip_runtime.h>
#include <hip/hip_bf16.h>
#include <math.h>

// ---------------------------------------------------------------------------
// StructuredStateTable on MI455X (gfx1250): bf16 WMMA GEMMs + WMMA GRU scan.
// Big GEMMs use GLOBAL_LOAD_ASYNC_TO_LDS_B128 double buffering (ASYNCcnt).
// ---------------------------------------------------------------------------

typedef __attribute__((ext_vector_type(16))) __bf16 v16bf;
typedef __attribute__((ext_vector_type(8)))  __bf16 v8bf;
typedef __attribute__((ext_vector_type(8)))  float  v8f;

#define BB 8
#define TT 1024
#define DD 768
#define NEE 8
#define DSS 128
#define HH 64

__device__ __forceinline__ v8f wmma_bf16(v16bf a, v16bf b, v8f c) {
  return __builtin_amdgcn_wmma_f32_16x16x32_bf16(false, a, false, b, (short)0, c,
                                                 false, false);
}

// A-operand (16xK tile, 16-bit): lane<16 -> row=lane, K-chunks {kb..kb+7, kb+16..kb+23}
// with kb = (lane>>4)*8.  Same mapping serves the B operand when B is stored [N][K]
// (row per output column).
__device__ __forceinline__ v16bf load_frag(const __bf16* base, int ld, int lane) {
  int r  = lane & 15;
  int kb = (lane >> 4) << 3;
  const __bf16* p = base + (size_t)r * ld + kb;
  v8bf lo = *(const v8bf*)(p);
  v8bf hi = *(const v8bf*)(p + 16);
  v16bf v;
#pragma unroll
  for (int i = 0; i < 8; ++i) { v[i] = lo[i]; v[i + 8] = hi[i]; }
  return v;
}

// Async DMA: global (16B per lane) -> LDS, tracked by ASYNCcnt.
__device__ __forceinline__ void async_load_b128(unsigned lds_off, const void* g) {
  asm volatile("global_load_async_to_lds_b128 %0, %1, off"
               :: "v"(lds_off), "v"(g) : "memory");
}
// Low 32 bits of a generic pointer to LDS are the DS byte address.
__device__ __forceinline__ unsigned lds_addr(const void* p) {
  return (unsigned)(uintptr_t)p;
}

// ---------------------------------------------------------------------------
// fp32 -> bf16 convert
// ---------------------------------------------------------------------------
__global__ void __launch_bounds__(256) cvt_bf16_kernel(__bf16* __restrict__ dst,
                                                       const float* __restrict__ src,
                                                       long n) {
  long i = (long)blockIdx.x * 256 + threadIdx.x;
  if (i < n) dst[i] = (__bf16)src[i];
}

// ---------------------------------------------------------------------------
// WMMA GEMM v1 (sync staging, 64x128 tile): C[M,N] = A[M,K] @ Bt[N,K]^T + bias
// Used for the skinny projected GEMM (N=128).
// ---------------------------------------------------------------------------
#define GBM 64
#define GBN 128
#define GKB 32

__global__ void __launch_bounds__(256) gemm_bf16_wmma(
    const __bf16* __restrict__ A, const __bf16* __restrict__ Bt,
    const float* __restrict__ bias, float* __restrict__ C,
    __bf16* __restrict__ Cbf, int M, int N, int K) {
  __shared__ __bf16 sA[GBM][GKB];
  __shared__ __bf16 sB[GBN][GKB];
  const int tid  = threadIdx.x;
  const int lane = tid & 31;
  const int wave = tid >> 5;
  const int wr   = wave >> 2;   // 0..1  (rows of 32)
  const int wc   = wave & 3;    // 0..3  (cols of 32)
  const int m0   = blockIdx.x * GBM;
  const int n0   = blockIdx.y * GBN;

  v8f acc[2][2] = {};

  for (int k0 = 0; k0 < K; k0 += GKB) {
    __syncthreads();
    {  // stage A tile (64x32): one 16B chunk per thread
      int row = tid >> 2, c = (tid & 3) << 3;
      *(v8bf*)&sA[row][c] = *(const v8bf*)&A[(size_t)(m0 + row) * K + k0 + c];
    }
#pragma unroll
    for (int i = 0; i < 2; ++i) {  // stage B tile (128x32): two chunks per thread
      int idx = tid + i * 256;
      int row = idx >> 2, c = (idx & 3) << 3;
      *(v8bf*)&sB[row][c] = *(const v8bf*)&Bt[(size_t)(n0 + row) * K + k0 + c];
    }
    __syncthreads();

    v16bf a0 = load_frag(&sA[wr * 32][0],      GKB, lane);
    v16bf a1 = load_frag(&sA[wr * 32 + 16][0], GKB, lane);
    v16bf b0 = load_frag(&sB[wc * 32][0],      GKB, lane);
    v16bf b1 = load_frag(&sB[wc * 32 + 16][0], GKB, lane);
    acc[0][0] = wmma_bf16(a0, b0, acc[0][0]);
    acc[0][1] = wmma_bf16(a0, b1, acc[0][1]);
    acc[1][0] = wmma_bf16(a1, b0, acc[1][0]);
    acc[1][1] = wmma_bf16(a1, b1, acc[1][1]);
  }

  const int cr = (lane >> 4) << 3;  // row offset 0 or 8
  const int cc = lane & 15;         // col within tile
#pragma unroll
  for (int ti = 0; ti < 2; ++ti)
#pragma unroll
    for (int tj = 0; tj < 2; ++tj) {
      int n   = n0 + wc * 32 + tj * 16 + cc;
      float b = bias ? bias[n] : 0.f;
#pragma unroll
      for (int i = 0; i < 8; ++i) {
        int m = m0 + wr * 32 + ti * 16 + cr + i;
        size_t off = (size_t)m * N + n;
        float v = acc[ti][tj][i] + b;
        C[off] = v;
        if (Cbf) Cbf[off] = (__bf16)v;
      }
    }
}

// ---------------------------------------------------------------------------
// WMMA GEMM v2 (async-to-LDS double buffered, 128x256 tile, 4x4 accs/wave).
// One ds_load_b128 per WMMA; DMA of panel k+1 overlaps WMMAs of panel k.
// ---------------------------------------------------------------------------
#define XBM 128
#define XBN 256
#define XKB 32

__global__ void __launch_bounds__(256) gemm_bf16_wmma_async(
    const __bf16* __restrict__ A, const __bf16* __restrict__ Bt,
    const float* __restrict__ bias, float* __restrict__ C,
    __bf16* __restrict__ Cbf, int M, int N, int K) {
  __shared__ __bf16 sA[2][XBM][XKB];   // 2 x 8 KB
  __shared__ __bf16 sB[2][XBN][XKB];   // 2 x 16 KB
  const int tid  = threadIdx.x;
  const int lane = tid & 31;
  const int wave = tid >> 5;
  const int wr   = wave >> 2;   // 0..1  (rows of 64)
  const int wc   = wave & 3;    // 0..3  (cols of 64)
  const int m0   = blockIdx.x * XBM;
  const int n0   = blockIdx.y * XBN;

  v8f acc[4][4] = {};

  // Per k-panel: A = 512 16B chunks (2/thread), B = 1024 chunks (4/thread)
  // -> 6 async instructions per wave per panel (ASYNCcnt += 6).
  auto issue_panel = [&](int buf, int k0) {
#pragma unroll
    for (int i = 0; i < 2; ++i) {
      int idx = tid + i * 256;
      int row = idx >> 2, c = (idx & 3) << 3;
      async_load_b128(lds_addr(&sA[buf][row][c]),
                      &A[(size_t)(m0 + row) * K + k0 + c]);
    }
#pragma unroll
    for (int i = 0; i < 4; ++i) {
      int idx = tid + i * 256;
      int row = idx >> 2, c = (idx & 3) << 3;
      async_load_b128(lds_addr(&sB[buf][row][c]),
                      &Bt[(size_t)(n0 + row) * K + k0 + c]);
    }
  };

  const int nk = K / XKB;
  issue_panel(0, 0);                        // prologue: fill buffer 0
  for (int ik = 0; ik < nk; ++ik) {
    __syncthreads();                        // all waves done reading buf[(ik+1)&1]
    if (ik + 1 < nk) {
      issue_panel((ik + 1) & 1, (ik + 1) * XKB);
      asm volatile("s_wait_asynccnt 6" ::: "memory");   // own current-panel DMA done
    } else {
      asm volatile("s_wait_asynccnt 0" ::: "memory");
    }
    __syncthreads();                        // everyone's current-panel data landed

    const int b = ik & 1;
    v16bf af[4], bf[4];
#pragma unroll
    for (int i = 0; i < 4; ++i)
      af[i] = load_frag(&sA[b][wr * 64 + i * 16][0], XKB, lane);
#pragma unroll
    for (int j = 0; j < 4; ++j)
      bf[j] = load_frag(&sB[b][wc * 64 + j * 16][0], XKB, lane);
#pragma unroll
    for (int i = 0; i < 4; ++i)
#pragma unroll
      for (int j = 0; j < 4; ++j)
        acc[i][j] = wmma_bf16(af[i], bf[j], acc[i][j]);
  }

  const int cr = (lane >> 4) << 3;
  const int cc = lane & 15;
#pragma unroll
  for (int ti = 0; ti < 4; ++ti)
#pragma unroll
    for (int tj = 0; tj < 4; ++tj) {
      int n   = n0 + wc * 64 + tj * 16 + cc;
      float b = bias ? bias[n] : 0.f;
#pragma unroll
      for (int i = 0; i < 8; ++i) {
        int m = m0 + wr * 64 + ti * 16 + cr + i;
        size_t off = (size_t)m * N + n;
        float v = acc[ti][tj][i] + b;
        C[off] = v;
        if (Cbf) Cbf[off] = (__bf16)v;
      }
    }
}

// ---------------------------------------------------------------------------
// Routing: logits = h.ek/scale, softmax over NE, routed = w * projected (bf16)
// ---------------------------------------------------------------------------
__global__ void __launch_bounds__(256) route_kernel(
    const float* __restrict__ h_seq, const float* __restrict__ ekeys,
    const float* __restrict__ proj, float* __restrict__ route_out,
    __bf16* __restrict__ routed_bf) {
  __shared__ float sl[NEE];
  const int row  = blockIdx.x;  // b*T + t
  const int tid  = threadIdx.x;
  const int lane = tid & 31;
  const int wave = tid >> 5;
  const float* h  = h_seq + (size_t)row * DD;
  const float* ek = ekeys + (size_t)wave * DD;

  float p = 0.f;
  for (int j = lane; j < DD; j += 32) p += h[j] * ek[j];
#pragma unroll
  for (int s = 16; s > 0; s >>= 1) p += __shfl_xor(p, s, 32);
  if (lane == 0) sl[wave] = p * 0.03608439182435161f;  // 1/sqrt(768)
  __syncthreads();

  float w[NEE], mx = -1e30f, sum = 0.f;
#pragma unroll
  for (int e = 0; e < NEE; ++e) mx = fmaxf(mx, sl[e]);
#pragma unroll
  for (int e = 0; e < NEE; ++e) { w[e] = __expf(sl[e] - mx); sum += w[e]; }
  float inv = 1.f / sum;
  if (tid < NEE) route_out[(size_t)row * NEE + tid] = w[tid] * inv;

  const float* pr = proj + (size_t)row * DSS;
  for (int idx = tid; idx < NEE * DSS; idx += 256) {
    int e = idx >> 7, d = idx & 127;
    routed_bf[((size_t)row * NEE + e) * DSS + d] = (__bf16)(w[e] * inv * pr[d]);
  }
}

// ---------------------------------------------------------------------------
// GRU scan: 4 blocks, each owns 16 of the 64 (b,e) recurrences.
// ---------------------------------------------------------------------------
__global__ void __launch_bounds__(256) gru_scan_kernel(
    const __bf16* __restrict__ routed_bf, const __bf16* __restrict__ Wih,
    const __bf16* __restrict__ Whh, const float* __restrict__ b_ih,
    const float* __restrict__ b_hh, const float* __restrict__ state0,
    float* __restrict__ state_out, __bf16* __restrict__ state_bf) {
  __shared__ __bf16 sX[16][DSS];
  __shared__ __bf16 sH[16][DSS];
  __shared__ float  fH[16][DSS];
  __shared__ float  sGi[16][3 * DSS];
  __shared__ float  sGh[16][3 * DSS];

  const int tid  = threadIdx.x;
  const int lane = tid & 31;
  const int wave = tid >> 5;
  const int r0   = blockIdx.x * 16;  // global row = b*NE + e

  for (int idx = tid; idx < 16 * DSS; idx += 256) {  // init hidden state
    int row = idx >> 7, d = idx & 127;
    int e = (r0 + row) & 7;
    float v = state0[e * DSS + d];
    fH[row][d] = v;
    sH[row][d] = (__bf16)v;
  }

  for (int t = 0; t < TT; ++t) {
    __syncthreads();
    {  // gather X_t [16,128] (one 16B chunk per thread)
      int row = tid >> 4, c = (tid & 15) << 3;
      int gr = r0 + row, b = gr >> 3, e = gr & 7;
      size_t src = (((size_t)(b * TT + t)) * NEE + e) * DSS + c;
      *(v8bf*)&sX[row][c] = *(const v8bf*)&routed_bf[src];
    }
    __syncthreads();

#pragma unroll
    for (int j = 0; j < 3; ++j) {  // wave handles N-tiles wave*3 .. +2 of 24
      int n0 = (wave * 3 + j) * 16;
      v8f gi = {}, gh = {};
#pragma unroll
      for (int kk = 0; kk < 4; ++kk) {
        int k0 = kk * 32;
        v16bf ax = load_frag(&sX[0][k0], DSS, lane);
        v16bf ah = load_frag(&sH[0][k0], DSS, lane);
        v16bf bi = load_frag(Wih + (size_t)n0 * DSS + k0, DSS, lane);
        v16bf bh = load_frag(Whh + (size_t)n0 * DSS + k0, DSS, lane);
        gi = wmma_bf16(ax, bi, gi);
        gh = wmma_bf16(ah, bh, gh);
      }
      int cr = (lane >> 4) << 3, cc = lane & 15;
#pragma unroll
      for (int i = 0; i < 8; ++i) {
        sGi[cr + i][n0 + cc] = gi[i];
        sGh[cr + i][n0 + cc] = gh[i];
      }
    }
    __syncthreads();

    for (int idx = tid; idx < 16 * DSS; idx += 256) {  // gate math (r,z,n)
      int row = idx >> 7, d = idx & 127;
      float ir = sGi[row][d]           + b_ih[d];
      float iz = sGi[row][DSS + d]     + b_ih[DSS + d];
      float in = sGi[row][2 * DSS + d] + b_ih[2 * DSS + d];
      float hr = sGh[row][d]           + b_hh[d];
      float hz = sGh[row][DSS + d]     + b_hh[DSS + d];
      float hn = sGh[row][2 * DSS + d] + b_hh[2 * DSS + d];
      float r = 1.f / (1.f + __expf(-(ir + hr)));
      float z = 1.f / (1.f + __expf(-(iz + hz)));
      float n = tanhf(in + r * hn);
      float hnew = (1.f - z) * n + z * fH[row][d];
      fH[row][d] = hnew;
      sH[row][d] = (__bf16)hnew;
      int gr = r0 + row, b = gr >> 3, e = gr & 7;
      size_t off = (((size_t)(b * TT + t)) * NEE + e) * DSS + d;
      state_out[off] = hnew;
      state_bf[off]  = (__bf16)hnew;
    }
  }
}

// ---------------------------------------------------------------------------
// Three scalar heads: DS->H (GELU exact) -> 1.  One 64-thread block per row.
// ---------------------------------------------------------------------------
__global__ void __launch_bounds__(64) heads_kernel(
    const float* __restrict__ state,
    const float* __restrict__ Wh1, const float* __restrict__ bh1,
    const float* __restrict__ Wh2, const float* __restrict__ bh2,
    const float* __restrict__ Wt1, const float* __restrict__ bt1,
    const float* __restrict__ Wt2, const float* __restrict__ bt2,
    const float* __restrict__ Wv1, const float* __restrict__ bv1,
    const float* __restrict__ Wv2, const float* __restrict__ bv2,
    float* __restrict__ holder, float* __restrict__ tagged,
    float* __restrict__ visible) {
  __shared__ float s[DSS];
  __shared__ float red[HH];
  const size_t row = blockIdx.x;
  const int tid = threadIdx.x;
  s[tid]      = state[row * DSS + tid];
  s[tid + 64] = state[row * DSS + 64 + tid];
  __syncthreads();

  const float* W1[3] = {Wh1, Wt1, Wv1};
  const float* B1[3] = {bh1, bt1, bv1};
  const float* W2[3] = {Wh2, Wt2, Wv2};
  const float* B2[3] = {bh2, bt2, bv2};
  float* O[3] = {holder, tagged, visible};

#pragma unroll
  for (int hh = 0; hh < 3; ++hh) {
    const float* w1 = W1[hh] + (size_t)tid * DSS;
    float acc = B1[hh][tid];
    for (int k = 0; k < DSS; ++k) acc = fmaf(w1[k], s[k], acc);
    float g = 0.5f * acc * (1.f + erff(acc * 0.7071067811865476f));
    red[tid] = g * W2[hh][tid];
    __syncthreads();
    for (int st = 32; st > 0; st >>= 1) {
      if (tid < st) red[tid] += red[tid + st];
      __syncthreads();
    }
    if (tid == 0) O[hh][row] = red[0] + B2[hh][0];
    __syncthreads();
  }
}

// ---------------------------------------------------------------------------
// Host-side orchestration
// ---------------------------------------------------------------------------
extern "C" void kernel_launch(void* const* d_in, const int* in_sizes, int n_in,
                              void* d_out, int out_size, void* d_ws, size_t ws_size,
                              hipStream_t stream) {
  const float* h_seq  = (const float*)d_in[0];
  const float* ekeys  = (const float*)d_in[1];
  const float* state0 = (const float*)d_in[2];
  const float* Wi     = (const float*)d_in[3];
  const float* bi     = (const float*)d_in[4];
  const float* W_ih   = (const float*)d_in[5];
  const float* W_hh   = (const float*)d_in[6];
  const float* b_ih   = (const float*)d_in[7];
  const float* b_hh   = (const float*)d_in[8];
  const float* Wm     = (const float*)d_in[9];
  const float* bm     = (const float*)d_in[10];
  const float* Ws     = (const float*)d_in[11];
  const float* bs     = (const float*)d_in[12];
  const float* Wh1 = (const float*)d_in[13]; const float* bh1 = (const float*)d_in[14];
  const float* Wh2 = (const float*)d_in[15]; const float* bh2 = (const float*)d_in[16];
  const float* Wt1 = (const float*)d_in[17]; const float* bt1 = (const float*)d_in[18];
  const float* Wt2 = (const float*)d_in[19]; const float* bt2 = (const float*)d_in[20];
  const float* Wv1 = (const float*)d_in[21]; const float* bv1 = (const float*)d_in[22];
  const float* Wv2 = (const float*)d_in[23]; const float* bv2 = (const float*)d_in[24];

  // Output layout (flat fp32, return order)
  float* out      = (float*)d_out;
  float* pooled   = out;                                  // [8,1024,768]
  float* memory   = out + 6291456L;                       // [8,1024,8,768]
  float* state    = out + 56623104L;                      // [8,1024,8,128]
  float* holder   = out + 65011712L;                      // [8,1024,8]
  float* tagged   = out + 65077248L;
  float* visible  = out + 65142784L;
  float* route_w  = out + 65208320L;

  // Workspace layout (byte offsets, 256B-aligned)
  char* ws = (char*)d_ws;
  __bf16* h_bf      = (__bf16*)(ws + 0);            // 6291456 el
  __bf16* Wi_bf     = (__bf16*)(ws + 12582912L);    // 98304 el  [128][768]
  __bf16* Wih_bf    = (__bf16*)(ws + 12779520L);    // 49152 el  [384][128]
  __bf16* Whh_bf    = (__bf16*)(ws + 12877824L);    // 49152 el
  __bf16* Wm_bf     = (__bf16*)(ws + 12976128L);    // 98304 el  [768][128]
  __bf16* Ws_bf     = (__bf16*)(ws + 13172736L);    // 4718592 el [768][6144]
  float*  proj_f32  = (float*) (ws + 22609920L);    // 1048576 el [8192,128]
  __bf16* routed_bf = (__bf16*)(ws + 26804224L);    // 8388608 el
  __bf16* state_bf  = (__bf16*)(ws + 43581440L);    // 8388608 el
  __bf16* mem_bf    = (__bf16*)(ws + 60358656L);    // 50331648 el

  auto cvt = [&](__bf16* dst, const float* src, long n) {
    cvt_bf16_kernel<<<dim3((unsigned)((n + 255) / 256)), dim3(256), 0, stream>>>(dst, src, n);
  };
  cvt(h_bf,   h_seq, 6291456L);
  cvt(Wi_bf,  Wi,    98304L);
  cvt(Wih_bf, W_ih,  49152L);
  cvt(Whh_bf, W_hh,  49152L);
  cvt(Wm_bf,  Wm,    98304L);
  cvt(Ws_bf,  Ws,    4718592L);

  // projected[8192,128] = h_bf[8192,768] @ Wi^T + bi (skinny N -> sync GEMM)
  gemm_bf16_wmma<<<dim3(8192 / GBM, 128 / GBN), dim3(256), 0, stream>>>(
      h_bf, Wi_bf, bi, proj_f32, nullptr, 8192, 128, 768);

  // routing softmax + routed (bf16)
  route_kernel<<<dim3(8192), dim3(256), 0, stream>>>(h_seq, ekeys, proj_f32,
                                                     route_w, routed_bf);

  // sequential GRU scan over T (writes state fp32 + bf16 mirror)
  gru_scan_kernel<<<dim3(4), dim3(256), 0, stream>>>(
      routed_bf, Wih_bf, Whh_bf, b_ih, b_hh, state0, state, state_bf);

  // memory[65536,768] = state_bf[65536,128] @ Wm^T + bm (fp32 + bf16 mirror)
  gemm_bf16_wmma_async<<<dim3(65536 / XBM, 768 / XBN), dim3(256), 0, stream>>>(
      state_bf, Wm_bf, bm, memory, mem_bf, 65536, 768, 128);

  // pooled[8192,768] = mem_bf[8192,6144] @ Ws^T + bs  (the 77 GF GEMM)
  gemm_bf16_wmma_async<<<dim3(8192 / XBM, 768 / XBN), dim3(256), 0, stream>>>(
      mem_bf, Ws_bf, bs, pooled, nullptr, 8192, 768, 6144);

  // scalar heads
  heads_kernel<<<dim3(65536), dim3(64), 0, stream>>>(
      state, Wh1, bh1, Wh2, bh2, Wt1, bt1, Wt2, bt2, Wv1, bv1, Wv2, bv2,
      holder, tagged, visible);
}
